// CfCLiquidCell_89807766159680
// MI455X (gfx1250) — matrix operand order repaired and checked by reading, not compiled, not run on hardware
//
#include <hip/hip_runtime.h>

// ---------------- types / helpers ----------------
typedef __attribute__((ext_vector_type(16))) __bf16 v16bf;
typedef __attribute__((ext_vector_type(8)))  float  v8f;
typedef __attribute__((ext_vector_type(8)))  unsigned short ushort8;
typedef __attribute__((ext_vector_type(16))) unsigned short ushort16;
typedef int int4g __attribute__((vector_size(16)));   // matches builtin param type

#define DEV __device__ __forceinline__

DEV unsigned short f2bf(float f) {
  unsigned int u = __builtin_bit_cast(unsigned int, f);
  u += 0x7FFFu + ((u >> 16) & 1u);            // round-to-nearest-even
  return (unsigned short)(u >> 16);
}
DEV float bf2f(unsigned short h) {
  unsigned int u = ((unsigned int)h) << 16;
  return __builtin_bit_cast(float, u);
}
DEV float sigmoidf_(float x) { return 1.0f / (1.0f + __expf(-x)); }
DEV float siluf_(float x)    { return x * sigmoidf_(x); }

// build a v16bf A/B fragment from two 16-byte chunks
DEV v16bf frag2x8(const unsigned short* p0, const unsigned short* p1) {
  ushort8 a = *(const ushort8*)p0;
  ushort8 b = *(const ushort8*)p1;
  ushort16 u;
#pragma unroll
  for (int i = 0; i < 8; ++i) { u[i] = a[i]; u[i + 8] = b[i]; }
  return __builtin_bit_cast(v16bf, u);
}

DEV v8f wmma_bf16(v16bf a, v16bf b, v8f c) {
  return __builtin_amdgcn_wmma_f32_16x16x32_bf16(false, a, false, b,
                                                 (short)0, c, false, false);
}

// ---------------- CDNA5 async global->LDS path ----------------
#if __has_builtin(__builtin_amdgcn_global_load_async_to_lds_b128)
#define HAVE_ASYNC_LDS 1
// builtin wants (int4 AS1* src, int4 AS3* dst, imm offset, imm cpol).
// generic LDS ptr: low 32 bits are the LDS offset (aperture rule) -> AS3 via int
#define TO_GLB(p) ((__attribute__((address_space(1))) int4g*)(unsigned long long)(p))
#define TO_LDS(p) ((__attribute__((address_space(3))) int4g*)(unsigned long long)(p))
#else
#define HAVE_ASYNC_LDS 0
#endif

DEV void wait_async0() {
#if __has_builtin(__builtin_amdgcn_s_wait_asynccnt)
  __builtin_amdgcn_s_wait_asynccnt(0);
#else
  asm volatile("s_wait_asynccnt 0x0" ::: "memory");
#endif
}

// ---------------- problem constants ----------------
#define Bn 8
#define Sn 2048
#define Hn 1024
#define NHn 16
#define HDn 64
#define NSn 64
#define Tn (Bn * Sn)           // 16384 tokens
#define Mh (Tn * NHn)          // 262144 head-rows

// ---------------- convert kernels ----------------
__global__ __launch_bounds__(256) void cvt_kernel(const float* __restrict__ s,
                                                  unsigned short* __restrict__ d) {
  int i = blockIdx.x * 256 + threadIdx.x;
  d[i] = f2bf(s[i]);
}

// src is (K x N) row-major fp32; dst is (N x K) row-major bf16 (transposed)
__global__ __launch_bounds__(256) void cvt_t_kernel(const float* __restrict__ s,
                                                    unsigned short* __restrict__ d,
                                                    int K, int N) {
  int i = blockIdx.x * 256 + threadIdx.x;
  int n = i / K, k = i - n * K;
  d[i] = f2bf(s[k * N + n]);
}

// ---------------- big GEMM: C(MxN) = A(MxK) * Bt(NxK)^T ----------------
#define TM 128
#define TN 128
#define TK 32
#define LSTR 40   // padded LDS stride (elements); 80B rows keep 16B alignment

__global__ __launch_bounds__(256) void gemm_bf16_128x128(
    const unsigned short* __restrict__ A,   // M x K bf16
    const unsigned short* __restrict__ Bt,  // N x K bf16 (B transposed)
    float* __restrict__ Cf,                 // f32 output (or null)
    unsigned short* __restrict__ Cb,        // bf16 output (or null)
    int M, int N, int K) {
  const int tid  = threadIdx.x;
  const int wid  = tid >> 5, lane = tid & 31;
  const int ln   = lane & 15, kh = lane >> 4;
  const int wm   = wid >> 1,  wn = wid & 1;          // 4 x 2 wave grid
  const int m0   = blockIdx.y * TM;
  const int n0   = blockIdx.x * TN;

  v8f acc[2][4];
#pragma unroll
  for (int i = 0; i < 2; ++i)
#pragma unroll
    for (int j = 0; j < 4; ++j) acc[i][j] = (v8f){0, 0, 0, 0, 0, 0, 0, 0};

#if HAVE_ASYNC_LDS
  // double-buffered tiles, filled by async DMA (ASYNCcnt-tracked)
  __shared__ unsigned short lsA[2][TM * LSTR];
  __shared__ unsigned short lsB[2][TN * LSTR];

  // each thread owns 2 A-chunks and 2 B-chunks of 16B per K-step
  const int i0 = tid >> 2,         j0 = (tid & 3) * 8;        // chunk c = tid
  const int i1 = (tid + 256) >> 2, j1 = (tid & 3) * 8;        // chunk c = tid+256

  auto issue = [&](int buf, int k0) {
    __builtin_amdgcn_global_load_async_to_lds_b128(
        TO_GLB(&A[(m0 + i0) * K + k0 + j0]), TO_LDS(&lsA[buf][i0 * LSTR + j0]), 0, 0);
    __builtin_amdgcn_global_load_async_to_lds_b128(
        TO_GLB(&A[(m0 + i1) * K + k0 + j1]), TO_LDS(&lsA[buf][i1 * LSTR + j1]), 0, 0);
    __builtin_amdgcn_global_load_async_to_lds_b128(
        TO_GLB(&Bt[(n0 + i0) * K + k0 + j0]), TO_LDS(&lsB[buf][i0 * LSTR + j0]), 0, 0);
    __builtin_amdgcn_global_load_async_to_lds_b128(
        TO_GLB(&Bt[(n0 + i1) * K + k0 + j1]), TO_LDS(&lsB[buf][i1 * LSTR + j1]), 0, 0);
  };

  issue(0, 0);
  int buf = 0;
  for (int k0 = 0; k0 < K; k0 += TK) {
    wait_async0();        // this wave's DMA for `buf` has landed in LDS
    __syncthreads();      // tile published; all waves done reading buf^1
    if (k0 + TK < K) issue(buf ^ 1, k0 + TK);   // overlap DMA with WMMAs

    v16bf bfr[4];
#pragma unroll
    for (int nt = 0; nt < 4; ++nt) {
      const unsigned short* p = &lsB[buf][(wn * 64 + nt * 16 + ln) * LSTR + kh * 16];
      bfr[nt] = frag2x8(p, p + 8);                    // B: k = e + 16*kh
    }
#pragma unroll
    for (int mt = 0; mt < 2; ++mt) {
      const unsigned short* p = &lsA[buf][(wm * 32 + mt * 16 + ln) * LSTR];
      v16bf af = frag2x8(p + kh * 16, p + kh * 16 + 8); // A: k = e + 16*kh
#pragma unroll
      for (int nt = 0; nt < 4; ++nt) acc[mt][nt] = wmma_bf16(af, bfr[nt], acc[mt][nt]);
    }
    buf ^= 1;
  }
#else
  // fallback: synchronous staging through VGPRs
  __shared__ unsigned short lsA[TM * LSTR];
  __shared__ unsigned short lsB[TN * LSTR];
  for (int k0 = 0; k0 < K; k0 += TK) {
#pragma unroll
    for (int c = tid; c < (TM * TK) / 8; c += 256) {
      int i = c >> 2, j = (c & 3) * 8;
      *(ushort8*)&lsA[i * LSTR + j] = *(const ushort8*)&A[(m0 + i) * K + k0 + j];
      *(ushort8*)&lsB[i * LSTR + j] = *(const ushort8*)&Bt[(n0 + i) * K + k0 + j];
    }
    if (k0 + TK < K) {
      __builtin_prefetch(&A[(m0 + (tid >> 1)) * K + k0 + TK], 0, 0);
      __builtin_prefetch(&Bt[(n0 + (tid >> 1)) * K + k0 + TK], 0, 0);
    }
    __syncthreads();
    v16bf bfr[4];
#pragma unroll
    for (int nt = 0; nt < 4; ++nt) {
      const unsigned short* p = &lsB[(wn * 64 + nt * 16 + ln) * LSTR + kh * 16];
      bfr[nt] = frag2x8(p, p + 8);
    }
#pragma unroll
    for (int mt = 0; mt < 2; ++mt) {
      const unsigned short* p = &lsA[(wm * 32 + mt * 16 + ln) * LSTR];
      v16bf af = frag2x8(p + kh * 16, p + kh * 16 + 8);
#pragma unroll
      for (int nt = 0; nt < 4; ++nt) acc[mt][nt] = wmma_bf16(af, bfr[nt], acc[mt][nt]);
    }
    __syncthreads();
  }
#endif

#pragma unroll
  for (int mt = 0; mt < 2; ++mt)
#pragma unroll
    for (int nt = 0; nt < 4; ++nt)
#pragma unroll
      for (int r = 0; r < 8; ++r) {
        int row = m0 + wm * 32 + mt * 16 + r + 8 * kh;
        int col = n0 + wn * 64 + nt * 16 + ln;
        float v = acc[mt][nt][r];
        if (Cb) Cb[row * N + col] = f2bf(v);
        else    Cf[row * N + col] = v;
      }
}

// ---------------- depthwise causal conv(K=4) + silu ----------------
__global__ __launch_bounds__(256) void conv_silu_kernel(
    const unsigned short* __restrict__ XZ,  // (T, 2H) bf16; x_path = cols [0,H)
    const float* __restrict__ cw,           // (H,1,4)
    const float* __restrict__ cb,           // (H)
    unsigned short* __restrict__ XP) {      // (T, H) bf16
  int idx = blockIdx.x * 256 + threadIdx.x;
  int t = idx >> 10, c = idx & (Hn - 1);
  int s = t & (Sn - 1);
  float acc = cb[c];
#pragma unroll
  for (int k = 0; k < 4; ++k) {
    int sp = s - 3 + k;
    if (sp >= 0) acc += cw[c * 4 + k] * bf2f(XZ[(t - 3 + k) * (2 * Hn) + c]);
  }
  XP[idx] = f2bf(siluf_(acc));
}

// ---------------- bb = silu(Xh @ bb_w + bb_b)  (M x 64 @ 64 x 64) ------------
__global__ __launch_bounds__(256) void bb_kernel(
    const unsigned short* __restrict__ A,   // (Mh, 64) bf16
    const unsigned short* __restrict__ Wt,  // (64, 64) bf16 transposed
    const float* __restrict__ bias,
    unsigned short* __restrict__ O) {       // (Mh, 64) bf16
  int wid = threadIdx.x >> 5, lane = threadIdx.x & 31;
  int ln = lane & 15, kh = lane >> 4;
  int tile = blockIdx.x * 8 + wid;
  int tm = tile >> 2, tn = tile & 3;
  int row = tm * 16 + ln, n0 = tn * 16;

  v8f acc = (v8f){0, 0, 0, 0, 0, 0, 0, 0};
#pragma unroll
  for (int kk = 0; kk < 2; ++kk) {
    int k0 = kk * 32;
    const unsigned short* pa = A + row * 64 + k0;
    const unsigned short* pb = Wt + (n0 + ln) * 64 + k0 + kh * 16;
    v16bf a = frag2x8(pa + kh * 16, pa + kh * 16 + 8);
    v16bf b = frag2x8(pb, pb + 8);
    acc = wmma_bf16(a, b, acc);
  }
  float bn = bias[n0 + ln];
#pragma unroll
  for (int r = 0; r < 8; ++r)
    O[(tm * 16 + r + 8 * kh) * 64 + n0 + ln] = f2bf(siluf_(acc[r] + bn));
}

// ------- 4 gate GEMMs fused -> candidate & decay (tanh/sigmoid epilogue) -----
__global__ __launch_bounds__(256) void gates_kernel(
    const unsigned short* __restrict__ BB,
    const unsigned short* __restrict__ Wf1, const unsigned short* __restrict__ Wf2,
    const unsigned short* __restrict__ Wta, const unsigned short* __restrict__ Wdc,
    const float* __restrict__ bf1, const float* __restrict__ bf2,
    const float* __restrict__ bta, const float* __restrict__ btb,
    const float* __restrict__ bdc,
    unsigned short* __restrict__ CAND, unsigned short* __restrict__ DEC) {
  int wid = threadIdx.x >> 5, lane = threadIdx.x & 31;
  int ln = lane & 15, kh = lane >> 4;
  int tile = blockIdx.x * 8 + wid;
  int tm = tile >> 2, tn = tile & 3;
  int row = tm * 16 + ln, n0 = tn * 16;

  v8f a1 = (v8f){0,0,0,0,0,0,0,0}, a2 = a1, a3 = a1, a4 = a1;
#pragma unroll
  for (int kk = 0; kk < 2; ++kk) {
    int k0 = kk * 32;
    const unsigned short* pa = BB + row * 64 + k0;
    v16bf a = frag2x8(pa + kh * 16, pa + kh * 16 + 8);
    int wo = (n0 + ln) * 64 + k0 + kh * 16;
    a1 = wmma_bf16(a, frag2x8(Wf1 + wo, Wf1 + wo + 8), a1);
    a2 = wmma_bf16(a, frag2x8(Wf2 + wo, Wf2 + wo + 8), a2);
    a3 = wmma_bf16(a, frag2x8(Wta + wo, Wta + wo + 8), a3);
    a4 = wmma_bf16(a, frag2x8(Wdc + wo, Wdc + wo + 8), a4);
  }
  int n = n0 + ln;
  float b1 = bf1[n], b2 = bf2[n], b3 = bta[n] + btb[n], b4 = bdc[n];
#pragma unroll
  for (int r = 0; r < 8; ++r) {
    float f1 = tanhf(a1[r] + b1);
    float f2 = tanhf(a2[r] + b2);
    float tau = sigmoidf_(a3[r] + b3);
    float dv = sigmoidf_(a4[r] + b4);
    int o = (tm * 16 + r + 8 * kh) * 64 + n;
    CAND[o] = f2bf(f1 * (1.0f - tau) + f2 * tau);
    DEC[o]  = f2bf(dv);
  }
}

// ---------------- sequential scan: h = d*h + (1-d)*c, in-place over CAND -----
__global__ __launch_bounds__(256) void scan_kernel(
    const unsigned short* __restrict__ DEC,
    unsigned short* __restrict__ CH) {     // candidate in, h_seq out (aliased)
  int tid = blockIdx.x * 256 + threadIdx.x;           // 8192 chains
  int base = (tid >> 10) * (Sn * NHn * NSn) + (tid & 1023);
  float h = 0.0f;
  for (int s = 0; s < Sn; ++s) {
    int i = base + s * (NHn * NSn);
    float d = bf2f(DEC[i]);
    float c = bf2f(CH[i]);
    h = d * h + (1.0f - d) * c;
    CH[i] = f2bf(h);
  }
}

// --------- out = (h_seq @ state_out_w + b) * silu(z_gate), re-laid to (T,H) --
__global__ __launch_bounds__(256) void stateout_kernel(
    const unsigned short* __restrict__ HS,   // (Mh, 64) bf16
    const unsigned short* __restrict__ Wt,   // (64, 64) bf16 transposed
    const float* __restrict__ bias,
    const unsigned short* __restrict__ XZ,   // (T, 2H); z_gate = cols [H,2H)
    unsigned short* __restrict__ O) {        // (T, H) bf16
  int wid = threadIdx.x >> 5, lane = threadIdx.x & 31;
  int ln = lane & 15, kh = lane >> 4;
  int tile = blockIdx.x * 8 + wid;
  int tm = tile >> 2, tn = tile & 3;        // tm == token index (NH==16)
  int row = tm * 16 + ln, n0 = tn * 16;

  v8f acc = (v8f){0, 0, 0, 0, 0, 0, 0, 0};
#pragma unroll
  for (int kk = 0; kk < 2; ++kk) {
    int k0 = kk * 32;
    const unsigned short* pa = HS + row * 64 + k0;
    const unsigned short* pb = Wt + (n0 + ln) * 64 + k0 + kh * 16;
    acc = wmma_bf16(frag2x8(pa + kh * 16, pa + kh * 16 + 8),
                    frag2x8(pb, pb + 8), acc);
  }
  float bn = bias[n0 + ln];
#pragma unroll
  for (int r = 0; r < 8; ++r) {
    int nh = r + 8 * kh;
    int col = nh * 64 + n0 + ln;
    float z = bf2f(XZ[tm * (2 * Hn) + Hn + col]);
    O[tm * Hn + col] = f2bf((acc[r] + bn) * siluf_(z));
  }
}

// ---------------- host-side launch ----------------
extern "C" void kernel_launch(void* const* d_in, const int* in_sizes, int n_in,
                              void* d_out, int out_size, void* d_ws, size_t ws_size,
                              hipStream_t stream) {
  const float* x      = (const float*)d_in[0];
  const float* inW    = (const float*)d_in[1];   // (1024, 2048)
  const float* convW  = (const float*)d_in[2];   // (1024,1,4)
  const float* convB  = (const float*)d_in[3];
  const float* bbW    = (const float*)d_in[4];   // (64,64)
  const float* bbB    = (const float*)d_in[5];
  const float* f1W    = (const float*)d_in[6];
  const float* f1B    = (const float*)d_in[7];
  const float* f2W    = (const float*)d_in[8];
  const float* f2B    = (const float*)d_in[9];
  const float* tauW   = (const float*)d_in[10];
  const float* tauB   = (const float*)d_in[11];
  const float* taub2  = (const float*)d_in[12];
  const float* decW   = (const float*)d_in[13];
  const float* decB   = (const float*)d_in[14];
  const float* soW    = (const float*)d_in[15];
  const float* soB    = (const float*)d_in[16];
  const float* outW   = (const float*)d_in[17];  // (1024, 1024)

  char* w = (char*)d_ws;
  // scratch layout (bytes); A_X is reused as XP, CAND as HS, DEC as OUT1
  unsigned short* A_X   = (unsigned short*)(w + 0);            // 16384x1024 bf16 (32 MB)
  unsigned short* WT_IN = (unsigned short*)(w + 33554432);     // 2048x1024 (4 MB)
  unsigned short* WT_OUT= (unsigned short*)(w + 37748736);     // 1024x1024 (2 MB)
  unsigned short* WT_BB = (unsigned short*)(w + 39845888);     // 64x64 x6
  unsigned short* WT_F1 = WT_BB + 4096;
  unsigned short* WT_F2 = WT_F1 + 4096;
  unsigned short* WT_TA = WT_F2 + 4096;
  unsigned short* WT_DC = WT_TA + 4096;
  unsigned short* WT_SO = WT_DC + 4096;
  unsigned short* XZ    = (unsigned short*)(w + 39895040);     // 16384x2048 (64 MB)
  unsigned short* BBb   = (unsigned short*)(w + 107003904);    // 262144x64 (32 MB)
  unsigned short* CAND  = (unsigned short*)(w + 140558336);    // 32 MB (-> h_seq)
  unsigned short* DEC   = (unsigned short*)(w + 174112768);    // 32 MB (-> gated out)
  unsigned short* XP    = A_X;     // conv output reuses x-bf16 buffer
  unsigned short* HS    = CAND;    // scan runs in place
  unsigned short* OUT1  = DEC;     // gated output reuses decay buffer

  // 1) converts
  cvt_kernel<<<(Tn * Hn) / 256, 256, 0, stream>>>(x, A_X);
  cvt_t_kernel<<<(1024 * 2048) / 256, 256, 0, stream>>>(inW, WT_IN, 1024, 2048);
  cvt_t_kernel<<<(1024 * 1024) / 256, 256, 0, stream>>>(outW, WT_OUT, 1024, 1024);
  cvt_t_kernel<<<16, 256, 0, stream>>>(bbW,  WT_BB, 64, 64);
  cvt_t_kernel<<<16, 256, 0, stream>>>(f1W,  WT_F1, 64, 64);
  cvt_t_kernel<<<16, 256, 0, stream>>>(f2W,  WT_F2, 64, 64);
  cvt_t_kernel<<<16, 256, 0, stream>>>(tauW, WT_TA, 64, 64);
  cvt_t_kernel<<<16, 256, 0, stream>>>(decW, WT_DC, 64, 64);
  cvt_t_kernel<<<16, 256, 0, stream>>>(soW,  WT_SO, 64, 64);

  // 2) xz = x @ in_proj_w   (16384x1024 @ 1024x2048) -> bf16
  gemm_bf16_128x128<<<dim3(2048 / TN, Tn / TM), 256, 0, stream>>>(
      A_X, WT_IN, nullptr, XZ, Tn, 2048, 1024);

  // 3) depthwise causal conv + silu  -> XP
  conv_silu_kernel<<<(Tn * Hn) / 256, 256, 0, stream>>>(XZ, convW, convB, XP);

  // 4) bb = silu(Xh @ bb_w + b)   (262144x64 @ 64x64)
  bb_kernel<<<(Mh / 16) * 4 / 8, 256, 0, stream>>>(XP, WT_BB, bbB, BBb);

  // 5) gates -> candidate, decay
  gates_kernel<<<(Mh / 16) * 4 / 8, 256, 0, stream>>>(
      BBb, WT_F1, WT_F2, WT_TA, WT_DC, f1B, f2B, tauB, taub2, decB, CAND, DEC);

  // 6) recurrence (in place: CAND becomes h_seq)
  scan_kernel<<<8192 / 256, 256, 0, stream>>>(DEC, CAND);

  // 7) state_out GEMM + bias, fused with silu(z_gate) multiply -> OUT1 (T,H)
  stateout_kernel<<<(Mh / 16) * 4 / 8, 256, 0, stream>>>(HS, WT_SO, soB, XZ, OUT1);

  // 8) final projection: d_out = OUT1 @ out_proj_w  (f32 output)
  gemm_bf16_128x128<<<dim3(1024 / TN, Tn / TM), 256, 0, stream>>>(
      OUT1, WT_OUT, (float*)d_out, nullptr, Tn, 1024, 1024);
}
